// GILayer_35261681500663
// MI455X (gfx1250) — matrix operand order repaired and checked by reading, not compiled, not run on hardware
//
#include <hip/hip_runtime.h>
#include <hip/hip_bf16.h>

#define EPSC       1e-7f
#define MIN_NORM   1e-15f
#define PROJ_EPS   1e-5f
#define NEG_SLOPE  0.2f

typedef __attribute__((ext_vector_type(16))) _Float16 v16h;
typedef __attribute__((ext_vector_type(8)))  float    v8f;
typedef __attribute__((ext_vector_type(4)))  unsigned u32x4;
typedef __attribute__((ext_vector_type(8)))  int      i32x8;
typedef __attribute__((ext_vector_type(4)))  int      i32x4;

#ifdef __has_builtin
#if __has_builtin(__builtin_amdgcn_tensor_load_to_lds)
#define USE_TDM 1
#endif
#endif

// ---------------- wave helpers (wave32) ----------------
__device__ inline float wsum(float v) {
  v += __shfl_xor(v, 1, 32);
  v += __shfl_xor(v, 2, 32);
  v += __shfl_xor(v, 4, 32);
  v += __shfl_xor(v, 8, 32);
  v += __shfl_xor(v, 16, 32);
  return v;
}
__device__ inline float gsum8(float v) {
  v += __shfl_xor(v, 1, 32);
  v += __shfl_xor(v, 2, 32);
  v += __shfl_xor(v, 4, 32);
  return v;
}

struct alignas(16) F4 { float x, y, z, w; };
__device__ inline F4 ld4(const float* p) { return *reinterpret_cast<const F4*>(p); }
__device__ inline void st4(float* p, F4 v) { *reinterpret_cast<F4*>(p) = v; }
__device__ inline F4 f4add(F4 a, F4 b) { return {a.x+b.x, a.y+b.y, a.z+b.z, a.w+b.w}; }
__device__ inline F4 f4sub(F4 a, F4 b) { return {a.x-b.x, a.y-b.y, a.z-b.z, a.w-b.w}; }
__device__ inline F4 f4scale(F4 a, float s) { return {a.x*s, a.y*s, a.z*s, a.w*s}; }
__device__ inline F4 f4axpby(float a, F4 x, float b, F4 y) {
  return {a*x.x + b*y.x, a*x.y + b*y.y, a*x.z + b*y.z, a*x.w + b*y.w};
}
__device__ inline float f4dot(F4 a, F4 b) { return a.x*b.x + a.y*b.y + a.z*b.z + a.w*b.w; }
__device__ inline F4 f4relu(F4 a) {
  return {fmaxf(a.x,0.f), fmaxf(a.y,0.f), fmaxf(a.z,0.f), fmaxf(a.w,0.f)};
}
__device__ inline F4 f4neg(F4 a) { return {-a.x, -a.y, -a.z, -a.w}; }

__device__ inline float wnorm(F4 v) { return sqrtf(wsum(f4dot(v, v))); }

__device__ inline float artanh_c(float x) {
  x = fminf(fmaxf(x, -1.f + EPSC), 1.f - EPSC);
  return 0.5f * logf((1.f + x) / (1.f - x));
}
__device__ inline F4 proj4(F4 v) {
  float n = fmaxf(wnorm(v), MIN_NORM);
  float maxn = 1.f - PROJ_EPS;
  if (n > maxn) v = f4scale(v, maxn / n);
  return v;
}
__device__ inline F4 expmap0_4(F4 u) {
  float n = fmaxf(wnorm(u), MIN_NORM);
  return f4scale(u, tanhf(n) / n);
}
__device__ inline F4 logmap0_4(F4 x) {
  float n = fmaxf(wnorm(x), MIN_NORM);
  return f4scale(x, artanh_c(n) / n);
}
__device__ inline F4 mobius_add4(F4 x, F4 y) {
  float x2 = wsum(f4dot(x, x));
  float y2 = wsum(f4dot(y, y));
  float xy = wsum(f4dot(x, y));
  float ca = 1.f + 2.f * xy + y2;
  float cb = 1.f - x2;
  float den = fmaxf(1.f + 2.f * xy + x2 * y2, MIN_NORM);
  return f4scale(f4axpby(ca, x, cb, y), 1.f / den);
}
__device__ inline float pdist4(F4 x, F4 y) {
  F4 ma = mobius_add4(f4neg(x), y);
  float n = sqrtf(wsum(f4dot(ma, ma)));
  return 2.f * artanh_c(n);
}

// order-preserving float<->uint for atomicMax-based segment max
__device__ inline unsigned f2ord(float f) {
  unsigned b = __float_as_uint(f);
  return (b & 0x80000000u) ? ~b : (b | 0x80000000u);
}
__device__ inline float ord2f(unsigned u) {
  return __uint_as_float((u & 0x80000000u) ? (u & 0x7fffffffu) : ~u);
}
__device__ inline bool finitef(float x) { return fabsf(x) <= 3.3e38f; }
__device__ inline float lrelu(float x) { return x > 0.f ? x : NEG_SLOPE * x; }

// ---------------- WMMA GEMM: Y[N,128] = X[N,128] @ W[128,128]^T (+bias) ----------------
// 256 threads = 8 waves. W (64KB fp32) staged to LDS by the Tensor Data Mover
// (one wave issues tensor_load_to_lds, waits TENSORcnt, block barrier).
// Each wave computes a 16x128 output tile: 8 N-tiles x 4 K-slices of
// v_wmma_f32_16x16x32_f16 with fp32 accumulation.
__global__ void gemm_wmma_128(const float* __restrict__ X, const float* __restrict__ W,
                              const float* __restrict__ bias, float* __restrict__ Y, int N) {
  __shared__ float Wf[128 * 128];  // 64KB
  int tid = threadIdx.x;
  int lane = tid & 31;
  int wv = tid >> 5;

#ifdef USE_TDM
  if (wv == 0) {
    unsigned long long ga = (unsigned long long)(uintptr_t)W;
    unsigned ldsa = (unsigned)(uintptr_t)(&Wf[0]);   // flat LDS addr low 32 bits = LDS offset
    // D# group0: count=1 | lds_addr | global_addr[31:0] | global_addr[56:32], type=2
    u32x4 g0 = { 1u, ldsa, (unsigned)ga,
                 (unsigned)((ga >> 32) & 0x01FFFFFFull) | (2u << 30) };
    // D# group1: mask=0, data_size=4B(code 2); tensor_dim0=16384; tensor_dim1=1;
    //            tile_dim0=16384; tile_dim1=1; tensor_dim0_stride=16384
    i32x8 g1 = { (int)(2u << 16),
                 (int)(16384u << 16),   // tensor_dim0[15:0] in bits 63:48
                 (int)(1u << 16),       // tensor_dim0 hi=0 | tensor_dim1 lo=1
                 (int)(16384u << 16),   // tensor_dim1 hi=0 | tile_dim0=16384
                 1,                     // tile_dim1=1, tile_dim2=0
                 16384,                 // tensor_dim0_stride lo32
                 0, 0 };
    i32x4 z4 = { 0, 0, 0, 0 };
    i32x8 z8 = { 0, 0, 0, 0, 0, 0, 0, 0 };
    __builtin_amdgcn_tensor_load_to_lds(g0, g1, z4, z4, z8, 0);
    __builtin_amdgcn_s_wait_tensorcnt(0);
  }
#else
  for (int i = tid; i < 128 * 128; i += 256) Wf[i] = W[i];
#endif
  __syncthreads();

  int row0 = (blockIdx.x * 8 + wv) * 16;
  if (row0 >= N) return;
  bool fullTile = (row0 + 16) <= N;

  int mrow = row0 + (lane & 15);
  const float* xr = X + (size_t)mrow * 128;
  int hi16 = (lane >= 16) ? 1 : 0;

  // A fragments: 16x32 f16 per K-slice; per ISA layout lanes 16-31 take K offset +8.
  // K-runs are 8 contiguous floats -> two aligned float4 loads each.
  v16h afrag[4];
  if (fullTile) {
    #pragma unroll
    for (int kt = 0; kt < 4; ++kt) {
      int kb = kt * 32 + (hi16 ? 8 : 0);
      F4 p0 = ld4(xr + kb);
      F4 p1 = ld4(xr + kb + 4);
      F4 p2 = ld4(xr + kb + 16);
      F4 p3 = ld4(xr + kb + 20);
      v16h a;
      a[0]  = (_Float16)p0.x; a[1]  = (_Float16)p0.y; a[2]  = (_Float16)p0.z; a[3]  = (_Float16)p0.w;
      a[4]  = (_Float16)p1.x; a[5]  = (_Float16)p1.y; a[6]  = (_Float16)p1.z; a[7]  = (_Float16)p1.w;
      a[8]  = (_Float16)p2.x; a[9]  = (_Float16)p2.y; a[10] = (_Float16)p2.z; a[11] = (_Float16)p2.w;
      a[12] = (_Float16)p3.x; a[13] = (_Float16)p3.y; a[14] = (_Float16)p3.z; a[15] = (_Float16)p3.w;
      afrag[kt] = a;
    }
  } else {
    bool valid = mrow < N;
    #pragma unroll
    for (int kt = 0; kt < 4; ++kt) {
      int kb = kt * 32 + (hi16 ? 8 : 0);
      v16h a;
      #pragma unroll
      for (int h = 0; h < 16; ++h) {
        int k = kb + (h < 8 ? h : h + 8);
        a[h] = valid ? (_Float16)xr[k] : (_Float16)0.0f;
      }
      afrag[kt] = a;
    }
  }

  #pragma unroll
  for (int nt = 0; nt < 8; ++nt) {
    v8f acc = {};
    int col = nt * 16 + (lane & 15);
    #pragma unroll
    for (int kt = 0; kt < 4; ++kt) {
      const float* wp = &Wf[(size_t)col * 128 + kt * 32 + (hi16 ? 8 : 0)];
      v16h b;
      #pragma unroll
      for (int h = 0; h < 16; ++h) b[h] = (_Float16)wp[h < 8 ? h : h + 8];
      acc = __builtin_amdgcn_wmma_f32_16x16x32_f16(false, afrag[kt], false, b,
                                                   (short)0, acc, false, false);
    }
    float bc = bias ? bias[col] : 0.0f;
    if (fullTile) {
      #pragma unroll
      for (int v = 0; v < 8; ++v) {
        int r = row0 + v + (hi16 ? 8 : 0);
        Y[(size_t)r * 128 + col] = acc[v] + bc;
      }
    } else {
      #pragma unroll
      for (int v = 0; v < 8; ++v) {
        int r = row0 + v + (hi16 ? 8 : 0);
        if (r < N) Y[(size_t)r * 128 + col] = acc[v] + bc;
      }
    }
  }
}

// ---------------- per-node attention dots (Euclidean) ----------------
__global__ void node_att(const float* __restrict__ x, const float* __restrict__ att,
                         float* __restrict__ ai, float* __restrict__ aj, int N) {
  int lane = threadIdx.x & 31;
  int n = blockIdx.x * 8 + (threadIdx.x >> 5);
  if (n >= N) return;
  F4 v = ld4(x + (size_t)n * 128 + lane * 4);
  int head = lane >> 3;
  int k0 = (lane & 7) * 4;
  const float* at = att + head * 64 + k0;
  const float* as = att + head * 64 + 32 + k0;
  float pt = v.x * at[0] + v.y * at[1] + v.z * at[2] + v.w * at[3];
  float ps = v.x * as[0] + v.y * as[1] + v.z * as[2] + v.w * as[3];
  pt = gsum8(pt);
  ps = gsum8(ps);
  if ((lane & 7) == 0) { ai[n * 4 + head] = pt; aj[n * 4 + head] = ps; }
}

// ---------------- per-node hyperbolic prep ----------------
__global__ void node_hyp(const float* __restrict__ x_h, float* __restrict__ mx_xh,
                         float* __restrict__ lx, const float* __restrict__ b_lin_h,
                         const float* __restrict__ att_h,
                         float* __restrict__ hi, float* __restrict__ hj, int N) {
  int lane = threadIdx.x & 31;
  int n = blockIdx.x * 8 + (threadIdx.x >> 5);
  if (n >= N) return;
  F4 x = ld4(x_h + (size_t)n * 128 + lane * 4);
  F4 mxv = ld4(mx_xh + (size_t)n * 128 + lane * 4);
  float n_x = fmaxf(wnorm(x), MIN_NORM);
  float n_mx = fmaxf(wnorm(mxv), MIN_NORM);
  F4 mv = f4scale(mxv, tanhf(n_mx / n_x * artanh_c(n_x)) / n_mx);
  mv = proj4(mv);
  F4 b = ld4(b_lin_h + lane * 4);
  F4 hb = proj4(expmap0_4(b));
  F4 xh = proj4(mobius_add4(mv, hb));
  st4(mx_xh + (size_t)n * 128 + lane * 4, xh);
  F4 l = logmap0_4(xh);
  st4(lx + (size_t)n * 128 + lane * 4, l);
  int head = lane >> 3;
  int k0 = (lane & 7) * 4;
  const float* at = att_h + head * 64 + k0;
  const float* as = att_h + head * 64 + 32 + k0;
  float pt = l.x * at[0] + l.y * at[1] + l.z * at[2] + l.w * at[3];
  float ps = l.x * as[0] + l.y * as[1] + l.z * as[2] + l.w * as[3];
  pt = gsum8(pt);
  ps = gsum8(ps);
  if ((lane & 7) == 0) { hi[n * 4 + head] = pt; hj[n * 4 + head] = ps; }
}

// ---------------- edge passes ----------------
__global__ void edge_passA(const float* __restrict__ xh, const int* __restrict__ ei,
                           const float* __restrict__ ai, const float* __restrict__ aj,
                           float* __restrict__ dval, unsigned* __restrict__ dmax,
                           float* __restrict__ eraw, unsigned* __restrict__ emax,
                           int E, int EP) {
  int lane = threadIdx.x & 31;
  int eid = blockIdx.x * 8 + (threadIdx.x >> 5);
  if (eid >= EP) return;
  int sn = (eid < E) ? ei[eid] : (eid - E);
  int dn = (eid < E) ? ei[E + eid] : (eid - E);
  F4 xd = ld4(xh + (size_t)dn * 128 + lane * 4);
  F4 xs = ld4(xh + (size_t)sn * 128 + lane * 4);
  float d = pdist4(xd, xs);
  if (lane == 0) {
    dval[eid] = d;
    atomicMax(&dmax[dn], f2ord(d));
  }
  if (lane < 4) {
    float r = lrelu(ai[dn * 4 + lane] + aj[sn * 4 + lane]);
    eraw[(size_t)eid * 4 + lane] = r;
    atomicMax(&emax[dn * 4 + lane], f2ord(r));
  }
}

__global__ void edge_passB(const int* __restrict__ ei, float* __restrict__ dval,
                           const unsigned* __restrict__ dmax, float* __restrict__ dsum,
                           float* __restrict__ eraw, const unsigned* __restrict__ emax,
                           float* __restrict__ esum, int E, int EP) {
  int eid = blockIdx.x * blockDim.x + threadIdx.x;
  if (eid >= EP) return;
  int dn = (eid < E) ? ei[E + eid] : (eid - E);
  float m = ord2f(dmax[dn]);
  if (!finitef(m)) m = 0.f;
  float e = expf(dval[eid] - m);
  dval[eid] = e;
  atomicAdd(&dsum[dn], e);
  #pragma unroll
  for (int h = 0; h < 4; ++h) {
    float mh = ord2f(emax[dn * 4 + h]);
    if (!finitef(mh)) mh = 0.f;
    float eh = expf(eraw[(size_t)eid * 4 + h] - mh);
    eraw[(size_t)eid * 4 + h] = eh;
    atomicAdd(&esum[dn * 4 + h], eh);
  }
}

__global__ void edge_passC(const int* __restrict__ ei, const float* __restrict__ xe,
                           const float* __restrict__ ed, const float* __restrict__ dsum,
                           const float* __restrict__ ee, const float* __restrict__ esum,
                           const float* __restrict__ hi, const float* __restrict__ hj,
                           float* __restrict__ agg_e, float* __restrict__ ahraw,
                           unsigned* __restrict__ hmax, int E, int EP) {
  int lane = threadIdx.x & 31;
  int eid = blockIdx.x * 8 + (threadIdx.x >> 5);
  if (eid >= EP) return;
  int sn = (eid < E) ? ei[eid] : (eid - E);
  int dn = (eid < E) ? ei[E + eid] : (eid - E);
  int h = lane >> 3;
  float al = ee[(size_t)eid * 4 + h] / (esum[dn * 4 + h] + 1e-16f);
  F4 v = ld4(xe + (size_t)sn * 128 + lane * 4);
  float* dst = agg_e + (size_t)dn * 128 + lane * 4;
  atomicAdd(dst + 0, v.x * al);
  atomicAdd(dst + 1, v.y * al);
  atomicAdd(dst + 2, v.z * al);
  atomicAdd(dst + 3, v.w * al);
  if (lane < 4) {
    float dnorm = ed[eid] / (dsum[dn] + 1e-16f);
    float r = lrelu((hi[dn * 4 + lane] + hj[sn * 4 + lane]) * dnorm);
    ahraw[(size_t)eid * 4 + lane] = r;
    atomicMax(&hmax[dn * 4 + lane], f2ord(r));
  }
}

__global__ void edge_passD(const int* __restrict__ ei, float* __restrict__ ahraw,
                           const unsigned* __restrict__ hmax, float* __restrict__ hsum,
                           int E, int EP) {
  int eid = blockIdx.x * blockDim.x + threadIdx.x;
  if (eid >= EP) return;
  int dn = (eid < E) ? ei[E + eid] : (eid - E);
  #pragma unroll
  for (int h = 0; h < 4; ++h) {
    float m = ord2f(hmax[dn * 4 + h]);
    if (!finitef(m)) m = 0.f;
    float e = expf(ahraw[(size_t)eid * 4 + h] - m);
    ahraw[(size_t)eid * 4 + h] = e;
    atomicAdd(&hsum[dn * 4 + h], e);
  }
}

__global__ void edge_passE(const int* __restrict__ ei, const float* __restrict__ lx,
                           const float* __restrict__ eh, const float* __restrict__ hsum,
                           float* __restrict__ agg_h, int E, int EP) {
  int lane = threadIdx.x & 31;
  int eid = blockIdx.x * 8 + (threadIdx.x >> 5);
  if (eid >= EP) return;
  int sn = (eid < E) ? ei[eid] : (eid - E);
  int dn = (eid < E) ? ei[E + eid] : (eid - E);
  int h = lane >> 3;
  float al = eh[(size_t)eid * 4 + h] / (hsum[dn * 4 + h] + 1e-16f);
  F4 v = ld4(lx + (size_t)sn * 128 + lane * 4);
  float* d = agg_h + (size_t)dn * 128 + lane * 4;
  atomicAdd(d + 0, v.x * al);
  atomicAdd(d + 1, v.y * al);
  atomicAdd(d + 2, v.z * al);
  atomicAdd(d + 3, v.w * al);
}

// ---------------- final fusion, one wave per node ----------------
__global__ void node_final(const float* __restrict__ agg_e, const float* __restrict__ agg_h,
                           const float* __restrict__ b_e, const float* __restrict__ b_h,
                           const float* __restrict__ att_hf, const float* __restrict__ att_ef,
                           float* __restrict__ out, int N) {
  int lane = threadIdx.x & 31;
  int n = blockIdx.x * 8 + (threadIdx.x >> 5);
  if (n >= N) return;
  F4 eo = f4relu(f4add(ld4(agg_e + (size_t)n * 128 + lane * 4), ld4(b_e + lane * 4)));
  F4 ot = f4relu(f4add(ld4(agg_h + (size_t)n * 128 + lane * 4), ld4(b_h + lane * 4)));
  F4 ho = proj4(expmap0_4(ot));
  F4 ye = proj4(expmap0_4(eo));
  float dist_f = pdist4(ho, ye) * att_hf[0];
  float nye = fmaxf(wnorm(ye), MIN_NORM);
  F4 xe2 = f4scale(ye, tanhf(dist_f * artanh_c(nye)) / nye);
  xe2 = proj4(xe2);
  F4 hf = proj4(mobius_add4(ho, xe2));
  F4 lh = logmap0_4(ho);
  F4 df = f4sub(lh, eo);
  float de = wsum(f4dot(df, df)) * att_ef[0];
  F4 ef = f4add(eo, f4scale(lh, de));
  st4(out + (size_t)n * 128 + lane * 4, hf);
  st4(out + (size_t)N * 128 + (size_t)n * 128 + lane * 4, ef);
}

// ---------------- host launch ----------------
extern "C" void kernel_launch(void* const* d_in, const int* in_sizes, int n_in,
                              void* d_out, int out_size, void* d_ws, size_t ws_size,
                              hipStream_t stream) {
  const float* x_e     = (const float*)d_in[0];
  const float* x_h     = (const float*)d_in[1];
  const int*   ei      = (const int*)d_in[2];
  const float* W_e     = (const float*)d_in[3];
  const float* b_lin_e = (const float*)d_in[4];
  const float* att_e   = (const float*)d_in[5];
  const float* b_e     = (const float*)d_in[6];
  const float* W_h     = (const float*)d_in[7];
  const float* b_lin_h = (const float*)d_in[8];
  const float* att_h   = (const float*)d_in[9];
  const float* b_h     = (const float*)d_in[10];
  const float* att_hf  = (const float*)d_in[11];
  const float* att_ef  = (const float*)d_in[12];

  int N  = in_sizes[0] / 128;
  int E  = in_sizes[2] / 2;
  int EP = E + N;
  size_t ND = (size_t)N * 128;

  float* ws    = (float*)d_ws;
  float* xe    = ws;
  float* xh    = ws + ND;          // mx then xh
  float* lx    = ws + 2 * ND;
  float* agg_e = ws + 3 * ND;
  float* agg_h = ws + 4 * ND;
  float* ai    = ws + 5 * ND;
  float* aj    = ai + 4 * (size_t)N;
  float* hi    = aj + 4 * (size_t)N;
  float* hj    = hi + 4 * (size_t)N;
  float* stats = hj + 4 * (size_t)N;  // dmax N | dsum N | emax 4N | esum 4N | hmax 4N | hsum 4N
  unsigned* dmax = (unsigned*)stats;
  float*    dsum = stats + (size_t)N;
  unsigned* emax = (unsigned*)(stats + 2 * (size_t)N);
  float*    esum = stats + 6 * (size_t)N;
  unsigned* hmax = (unsigned*)(stats + 10 * (size_t)N);
  float*    hsum = stats + 14 * (size_t)N;
  float* dval  = stats + 18 * (size_t)N;   // EP   (becomes exp(d))
  float* eraw  = dval + (size_t)EP;        // EP*4 (becomes exp(alpha_e))
  float* ahraw = eraw + 4 * (size_t)EP;    // EP*4 (becomes exp(alpha_h))

  (void)hipMemsetAsync(agg_e, 0, 2 * ND * sizeof(float), stream);
  (void)hipMemsetAsync(stats, 0, 18 * (size_t)N * sizeof(float), stream);

  dim3 blk(256);
  int gGemm = (N + 127) / 128;
  int gN8 = (N + 7) / 8;
  int gE8 = (EP + 7) / 8;
  int gE256 = (EP + 255) / 256;

  gemm_wmma_128<<<gGemm, blk, 0, stream>>>(x_e, W_e, b_lin_e, xe, N);
  gemm_wmma_128<<<gGemm, blk, 0, stream>>>(x_h, W_h, nullptr, xh, N);
  node_att<<<gN8, blk, 0, stream>>>(xe, att_e, ai, aj, N);
  node_hyp<<<gN8, blk, 0, stream>>>(x_h, xh, lx, b_lin_h, att_h, hi, hj, N);
  edge_passA<<<gE8, blk, 0, stream>>>(xh, ei, ai, aj, dval, dmax, eraw, emax, E, EP);
  edge_passB<<<gE256, blk, 0, stream>>>(ei, dval, dmax, dsum, eraw, emax, esum, E, EP);
  edge_passC<<<gE8, blk, 0, stream>>>(ei, xe, dval, dsum, eraw, esum, hi, hj, agg_e, ahraw, hmax, E, EP);
  edge_passD<<<gE256, blk, 0, stream>>>(ei, ahraw, hmax, hsum, E, EP);
  edge_passE<<<gE8, blk, 0, stream>>>(ei, lx, ahraw, hsum, agg_h, E, EP);
  node_final<<<gN8, blk, 0, stream>>>(agg_e, agg_h, b_e, b_h, att_hf, att_ef, (float*)d_out, N);
}
